// SjSTDPConv_21698174780269
// MI455X (gfx1250) — matrix operand order repaired and checked by reading, not compiled, not run on hardware
//
#include <hip/hip_runtime.h>

typedef __attribute__((ext_vector_type(16))) _Float16 v16h;
typedef __attribute__((ext_vector_type(8)))  float    v8f;

#define T_    32
#define N_    16
#define CIN_  2
#define COUT_ 8
#define H_    128
#define W_    128
#define K_    5
#define PAD_  2
#define HP_   (H_ + 2 * PAD_)   // 132
#define WP_   (W_ + 2 * PAD_)   // 132
#define KK_   (CIN_ * K_ * K_)  // 50
#define NW_   (COUT_ * KK_)     // 400
#define XROW_ 144               // LDS row pitch (132 used)

__device__ __forceinline__ float lif_inv_tau() { return 1.0f / 100.0f; }
__device__ __forceinline__ float trace_decay() { return 1.0f - 1.0f / 30.0f; }

// conv B window base for patch index k (compile-time folded under full unroll)
__device__ __forceinline__ constexpr int bbase_conv(int k) {
    return (k < KK_) ? (((k / 25) * K_ + ((k % 25) / 5)) * XROW_ + ((k % 25) % 5))
                     : (CIN_ * K_ * XROW_);              // zero row
}

// aligned 16-byte LDS load into half of a v16h (seg 0: e0..7, seg 1: e8..15)
__device__ __forceinline__ void load8h(v16h* dst, int seg, const _Float16* p) {
    ((uint4*)dst)[seg] = *(const uint4*)p;               // 16B-aligned ds_load_b128
}

// ---------------------------------------------------------------------------
// Init: zero v / tr_post / tr_pre, copy weights into workspace (f32 master)
// ---------------------------------------------------------------------------
__global__ void init_state(const float* __restrict__ w_in, float* __restrict__ w,
                           float* __restrict__ v, float* __restrict__ tr_post,
                           float* __restrict__ tr_pre) {
    int i = blockIdx.x * blockDim.x + threadIdx.x;
    if (i < NW_) w[i] = w_in[i];
    if (i < N_ * COUT_ * H_ * W_) { v[i] = 0.0f; tr_post[i] = 0.0f; }
    if (i < N_ * CIN_ * HP_ * WP_) tr_pre[i] = 0.0f;
}

// ---------------------------------------------------------------------------
// Conv (implicit GEMM, v_wmma_f32_16x16x32_f16) + LIF + spike + tr_post.
// Block = one (batch, row); 8 waves x 16 px. A = weights 16x64 (ds_load_b128
// fragments), B = im2col gathered with compile-time window bases (cndmask+add
// per element, zero-row for padded K). No divergent index math.
// ---------------------------------------------------------------------------
__global__ void conv_lif(const float* __restrict__ x_t, const float* __restrict__ x_next,
                         const float* __restrict__ w,
                         float* __restrict__ v, float* __restrict__ tr_post,
                         float* __restrict__ out_t) {
    __shared__ __align__(16) _Float16 sW[16 * 64];                 // padded A tile
    __shared__ __align__(16) _Float16 sX[CIN_ * K_ * XROW_ + XROW_]; // window + zero row
    const int XZ = CIN_ * K_ * XROW_;

    const int tid = threadIdx.x;
    const int b = blockIdx.x / H_;
    const int y = blockIdx.x % H_;

    for (int i = tid; i < 16 * 64; i += 256) {
        int r = i >> 6, k = i & 63;
        float wv = (r < COUT_ && k < KK_) ? w[r * KK_ + k] : 0.0f;
        sW[i] = (_Float16)wv;
    }
    for (int i = tid; i < CIN_ * K_ * 132; i += 256) {
        int xx = i % 132;
        int kh = (i / 132) % K_;
        int ci = i / (132 * K_);
        int yy = y + kh;                                           // padded row coord
        float xv = 0.0f;
        if (yy >= PAD_ && yy < H_ + PAD_ && xx >= PAD_ && xx < W_ + PAD_)
            xv = x_t[((b * CIN_ + ci) * H_ + (yy - PAD_)) * W_ + (xx - PAD_)];
        sX[(ci * K_ + kh) * XROW_ + xx] = (_Float16)xv;
    }
    if (tid < XROW_) sX[XZ + tid] = (_Float16)0.0f;                // zero row
    if (tid < 8) {                                                 // prefetch x_{t+1}
        int ci = tid >> 2, q = tid & 3;
        __builtin_prefetch(x_next + ((size_t)(b * CIN_ + ci) * H_ + y) * W_ + q * 32, 0, 1);
    }
    __syncthreads();

    const int lane = tid & 31;
    const int x0 = (tid >> 5) * 16;
    const int rc = lane & 15;
    const int hs = lane >> 4;
    const int xr = x0 + rc;                                        // B pixel column

    v8f c = {};
#pragma unroll
    for (int kt = 0; kt < 2; ++kt) {
        v16h a;
        load8h(&a, 0, &sW[rc * 64 + kt * 32 + 16 * hs]);
        load8h(&a, 1, &sW[rc * 64 + kt * 32 + 16 * hs + 8]);
        v16h bm;
#pragma unroll
        for (int e = 0; e < 16; ++e) {
            const int base0 = bbase_conv(kt * 32 + e);             // hs == 0 (immediate)
            const int base1 = bbase_conv(kt * 32 + e + 16);        // hs == 1 (immediate)
            bm[e] = sX[(hs ? base1 : base0) + xr];
        }
        c = __builtin_amdgcn_wmma_f32_16x16x32_f16(false, a, false, bm,
                                                   (short)0, c, false, false);
    }

    if (lane < 16) {                                               // M=0..7 valid rows
        int px = x0 + lane;
#pragma unroll
        for (int r = 0; r < COUT_; ++r) {
            int idx  = ((b * COUT_ + r) * H_ + y) * W_ + px;
            float vv = v[idx];
            vv += (c[r] - vv) * lif_inv_tau();
            float sp = (vv >= 1.0f) ? 1.0f : 0.0f;
            v[idx]       = vv * (1.0f - sp);
            out_t[idx]   = sp;
            tr_post[idx] = tr_post[idx] * trace_decay() + sp;
        }
    }
}

// ---------------------------------------------------------------------------
// tr_pre update on the padded grid + zero dw accumulator
// ---------------------------------------------------------------------------
__global__ void trpre_update(const float* __restrict__ x_t, float* __restrict__ tr_pre,
                             float* __restrict__ dw) {
    int i = blockIdx.x * blockDim.x + threadIdx.x;
    if (i < NW_) dw[i] = 0.0f;
    int total = N_ * CIN_ * HP_ * WP_;
    if (i >= total) return;
    int xx = i % WP_;
    int t2 = i / WP_;
    int yy = t2 % HP_;
    int bc = t2 / HP_;
    float xv = 0.0f;
    if (yy >= PAD_ && yy < H_ + PAD_ && xx >= PAD_ && xx < W_ + PAD_)
        xv = x_t[(bc * H_ + (yy - PAD_)) * W_ + (xx - PAD_)];
    tr_pre[i] = tr_pre[i] * trace_decay() + xv;
}

// ---------------------------------------------------------------------------
// STDP: C[o,ki] += tr_post*x_pad + spike*tr_pre over pixels (WMMA, K=pixels).
// Block = (batch, 2 rows); 8 waves x 32-px chunks. ki->window-base via a
// 64-entry LDS table (1 read/tile/lane); B elements are consecutive halves
// (immediate-offset ds_load_u16 run); A fragments are 2x ds_load_b128.
// ---------------------------------------------------------------------------
__global__ void stdp_corr(const float* __restrict__ x_t, const float* __restrict__ tr_pre,
                          const float* __restrict__ tr_post, const float* __restrict__ spike_t,
                          float* __restrict__ dw) {
    __shared__ __align__(16) _Float16 sXp[CIN_ * 6 * XROW_ + 2 * XROW_];
    __shared__ __align__(16) _Float16 sTp[CIN_ * 6 * XROW_ + 2 * XROW_];
    __shared__ __align__(16) _Float16 sPost[COUT_ * 2 * 128 + 160];
    __shared__ __align__(16) _Float16 sSpk[COUT_ * 2 * 128 + 160];
    __shared__ unsigned short sKB[64];                   // ki -> window base
    __shared__ float acc[NW_];
    const int XZ2 = CIN_ * 6 * XROW_;                    // zero region base (288 wide)
    const int PZ  = COUT_ * 2 * 128;                     // zero region base (160 wide)

    const int tid = threadIdx.x;
    const int b  = blockIdx.x / (H_ / 2);
    const int y0 = (blockIdx.x % (H_ / 2)) * 2;

    for (int i = tid; i < NW_; i += 256) acc[i] = 0.0f;
    if (tid < 64) {
        int ki = tid, base = XZ2;
        if (ki < KK_) {
            int ci = ki / 25, r5 = ki % 25, kh = r5 / 5, kw = r5 % 5;
            base = (ci * 6 + kh) * XROW_ + kw;
        }
        sKB[tid] = (unsigned short)base;
    }
    for (int i = tid; i < CIN_ * 6 * 132; i += 256) {
        int xx = i % 132;
        int r  = (i / 132) % 6;
        int ci = i / (132 * 6);
        int yy = y0 + r;                                 // padded coord (<= 131)
        float xv = 0.0f;
        if (yy >= PAD_ && yy < H_ + PAD_ && xx >= PAD_ && xx < W_ + PAD_)
            xv = x_t[((b * CIN_ + ci) * H_ + (yy - PAD_)) * W_ + (xx - PAD_)];
        int o = (ci * 6 + r) * XROW_ + xx;
        sXp[o] = (_Float16)xv;
        sTp[o] = (_Float16)tr_pre[((b * CIN_ + ci) * HP_ + yy) * WP_ + xx];
    }
    for (int i = tid; i < 2 * XROW_; i += 256) {         // zero regions
        sXp[XZ2 + i] = (_Float16)0.0f;
        sTp[XZ2 + i] = (_Float16)0.0f;
    }
    for (int i = tid; i < COUT_ * 2 * 128; i += 256) {
        int xx = i % 128;
        int r  = (i / 128) % 2;
        int o  = i / (128 * 2);
        int idx = ((b * COUT_ + o) * H_ + y0 + r) * W_ + xx;
        sPost[(o * 2 + r) * 128 + xx] = (_Float16)tr_post[idx];
        sSpk [(o * 2 + r) * 128 + xx] = (_Float16)spike_t[idx];
    }
    if (tid < 160) { sPost[PZ + tid] = (_Float16)0.0f; sSpk[PZ + tid] = (_Float16)0.0f; }
    __syncthreads();

    const int lane = tid & 31;
    const int wv   = tid >> 5;
    const int row  = wv >> 2;                            // 0..1
    const int x0   = (wv & 3) * 32;                      // 32-px chunk
    const int rc   = lane & 15;
    const int hs   = lane >> 4;

    // A fragments (K = 32 pixels): 2x aligned ds_load_b128 each
    const int basep = (rc < COUT_) ? ((rc * 2 + row) * 128) : PZ;
    v16h a_post, a_spk;
    load8h(&a_post, 0, &sPost[basep + x0 + 16 * hs]);
    load8h(&a_post, 1, &sPost[basep + x0 + 16 * hs + 8]);
    load8h(&a_spk,  0, &sSpk [basep + x0 + 16 * hs]);
    load8h(&a_spk,  1, &sSpk [basep + x0 + 16 * hs + 8]);

    v8f csum[4];
#pragma unroll
    for (int tile = 0; tile < 4; ++tile) {
        const int base_t = (int)sKB[tile * 16 + rc] + row * XROW_ + x0 + 16 * hs;
        v16h b_x, b_tr;
#pragma unroll
        for (int e = 0; e < 16; ++e) {                   // consecutive halves
            b_x[e]  = sXp[base_t + e];
            b_tr[e] = sTp[base_t + e];
        }
        v8f c = {};
        c = __builtin_amdgcn_wmma_f32_16x16x32_f16(false, a_post, false, b_x,
                                                   (short)0, c, false, false);
        c = __builtin_amdgcn_wmma_f32_16x16x32_f16(false, a_spk, false, b_tr,
                                                   (short)0, c, false, false);
        csum[tile] = c;
    }

    if (lane < 16) {
#pragma unroll
        for (int tile = 0; tile < 4; ++tile) {
            int ki = tile * 16 + lane;
            if (ki < KK_) {
#pragma unroll
                for (int r = 0; r < COUT_; ++r)
                    atomicAdd(&acc[r * KK_ + ki], csum[tile][r]);
            }
        }
    }
    __syncthreads();
    for (int i = tid; i < NW_; i += 256)
        atomicAdd(&dw[i], acc[i]);
}

// ---------------------------------------------------------------------------
// w += -ETA * dw
// ---------------------------------------------------------------------------
__global__ void apply_dw(float* __restrict__ w, const float* __restrict__ dw) {
    int i = blockIdx.x * blockDim.x + threadIdx.x;
    if (i < NW_) w[i] += -0.03f * dw[i];
}

extern "C" void kernel_launch(void* const* d_in, const int* in_sizes, int n_in,
                              void* d_out, int out_size, void* d_ws, size_t ws_size,
                              hipStream_t stream) {
    const float* x_seq = (const float*)d_in[0];   // [T,N,CIN,H,W]
    const float* w_in  = (const float*)d_in[1];   // [COUT,CIN,K,K]
    float* out = (float*)d_out;                   // [T,N,COUT,H,W]

    float* ws      = (float*)d_ws;
    float* w       = ws;
    float* dw      = ws + 512;
    float* v       = ws + 1024;
    float* tr_post = v + (size_t)N_ * COUT_ * H_ * W_;
    float* tr_pre  = tr_post + (size_t)N_ * COUT_ * H_ * W_;

    init_state<<<(N_ * COUT_ * H_ * W_) / 256, 256, 0, stream>>>(w_in, w, v, tr_post, tr_pre);

    const size_t xstep = (size_t)N_ * CIN_ * H_ * W_;
    for (int t = 0; t < T_; ++t) {
        const float* x_t   = x_seq + (size_t)t * xstep;
        const float* x_nx  = (t + 1 < T_) ? x_t + xstep : x_t;
        float*       out_t = out + (size_t)t * N_ * COUT_ * H_ * W_;
        conv_lif<<<N_ * H_, 256, 0, stream>>>(x_t, x_nx, w, v, tr_post, out_t);
        trpre_update<<<(N_ * CIN_ * HP_ * WP_ + 255) / 256, 256, 0, stream>>>(x_t, tr_pre, dw);
        stdp_corr<<<N_ * (H_ / 2), 256, 0, stream>>>(x_t, tr_pre, tr_post, out_t, dw);
        apply_dw<<<2, 256, 0, stream>>>(w, dw);
    }
}
